// NonAutoregressiveMultiGNNv1_75419625718341
// MI455X (gfx1250) — matrix operand order repaired and checked by reading, not compiled, not run on hardware
//
#include <hip/hip_runtime.h>
#include <hip/hip_fp16.h>
#include <math.h>

typedef __attribute__((ext_vector_type(16))) _Float16 v16h;
typedef __attribute__((ext_vector_type(8)))  float    v8f;

// ---------------- problem sizes (fixed by reference) ----------------
static constexpr int Nn   = 8000;
static constexpr int Ee   = 128000;
static constexpr int Cf   = 2;
static constexpr int Mens = 3;
static constexpr int Ly   = 5;
static constexpr int NCr  = Nn * Cf;      // 16000 node rows (mult of 16)
static constexpr int ECr  = Ee * Cf;      // 256000 edge rows (mult of 16)
static constexpr int NSI  = 78;           // 64 + 14 engineered features
static constexpr int NSIP = 80;           // padded K for WMMA
static constexpr int NSH  = 128;
static constexpr int HEADS= 4, HC = 32, OUTD = 4;

__device__ __forceinline__ float sigmoidf_(float x){ return 1.f/(1.f+__expf(-x)); }
__device__ __forceinline__ float siluf_(float x){ return x*sigmoidf_(x); }

// =====================================================================
// Compile-time-shaped wave-level WMMA GEMM.
// Out[M x N] = act(A[M x K] @ W[K x N] + b); one wave = one 16x16 tile.
// K, N compile-time; M must be a multiple of 16 (true for all call sites),
// so there are NO runtime bounds checks on the load path: A is read as
// float4 pairs (b128), B as 16 strided b32 loads, K-loop fully unrolled
// into back-to-back v_wmma_f32_16x16x32_f16.
// scat != null: rows are edge rows r=e*2+c -> atomicAdd into row dst[e]*2+c
// (fused edge-message GEMM + scatter-sum; avoids the 131MB ce_s tensor).
// =====================================================================
template<int K, int N>
__global__ void k_gemmT(const float* __restrict__ A, const float* __restrict__ W,
                        const float* __restrict__ bias, float* __restrict__ Out,
                        const int* __restrict__ scat, int Mrows, int act)
{
  constexpr int NT = N >> 4;
  constexpr int KC = (K + 31) >> 5;
  const int lane = threadIdx.x & 31;
  const int wid  = blockIdx.x * (blockDim.x >> 5) + (threadIdx.x >> 5);
  const int mt = wid / NT, nt = wid - mt * NT;
  const int row0 = mt << 4;
  if (row0 >= Mrows) return;               // wave-uniform exit (grid overshoot)
  const int g  = lane >> 4;                // lane group (0/1)
  const int hl = lane & 15;
  const float* Ar = A + (size_t)(row0 + hl) * K;
  const int bcol = (nt << 4) + hl;
  v8f acc = {};
  #pragma unroll
  for (int c = 0; c < KC; ++c) {
    const int k0 = c << 5;
    v16h a = {}; v16h b = {};
    // A 16x32 f16 layout: lane(g,m), element e in [8*grp, 8*grp+8):
    //   K = k0 + 16*grp + 8*g + (e - 8*grp)  -> 8 consecutive floats = 2x float4
    #pragma unroll
    for (int grp = 0; grp < 2; ++grp) {
      if (k0 + 16 * grp + 16 <= K) {       // compile-time (K mult of 16)
        const int kb = k0 + 16 * grp + 8 * g;
        const float4 f0 = *(const float4*)(Ar + kb);
        const float4 f1 = *(const float4*)(Ar + kb + 4);
        a[8*grp+0]=(_Float16)f0.x; a[8*grp+1]=(_Float16)f0.y;
        a[8*grp+2]=(_Float16)f0.z; a[8*grp+3]=(_Float16)f0.w;
        a[8*grp+4]=(_Float16)f1.x; a[8*grp+5]=(_Float16)f1.y;
        a[8*grp+6]=(_Float16)f1.z; a[8*grp+7]=(_Float16)f1.w;
      }
    }
    // B 32x16 f16 layout: lane(g,n): element e holds K = k0 + 8g + e + 8*(e>>3), col n
    {
      const float* Wr = W + (size_t)(k0 + 8 * g) * N + bcol;
      #pragma unroll
      for (int e = 0; e < 16; ++e)
        if (e < 8 || k0 + 32 <= K) b[e] = (_Float16)Wr[(size_t)(e + (e >> 3) * 8) * N];
    }
    acc = __builtin_amdgcn_wmma_f32_16x16x32_f16(false, a, false, b,
                                                 (short)0, acc, false, false);
  }
  // C/D layout: VGPR j -> row = row0 + j + 8g, col = bcol
  float bb = bias ? bias[bcol] : 0.f;
  #pragma unroll
  for (int j = 0; j < 8; ++j) {
    const int row = row0 + j + 8 * g;
    float v = acc[j] + bb;
    if (act == 1) v = siluf_(v);
    if (scat) {
      const int orow = scat[row >> 1] * 2 + (row & 1);
      atomicAdd(&Out[(size_t)orow * N + bcol], v);
    } else {
      Out[(size_t)row * N + bcol] = v;
    }
  }
}

// zero-pad Wv_Ws [3][78][128] -> [3][80][128] so the K=80 GEMM never reads OOB
__global__ void k_padwv(const float* __restrict__ W, float* __restrict__ Wp)
{
  int i = blockIdx.x * blockDim.x + threadIdx.x;
  if (i >= Mens * NSIP * NSH) return;
  int m = i / (NSIP * NSH);
  int r = (i / NSH) % NSIP;
  int c = i % NSH;
  Wp[i] = (r < NSI) ? W[((size_t)m * NSI + r) * NSH + c] : 0.f;
}

// ---------------- LayerNorm over D dims (D <= 128), one block/row ----------------
// writes stride-Dpad output, zero-filling columns [D, Dpad)
__global__ void k_layernorm(const float* __restrict__ x, const float* __restrict__ g,
                            const float* __restrict__ b, float* __restrict__ y,
                            int D, int Dpad)
{
  __shared__ float red[128];
  int row = blockIdx.x, t = threadIdx.x;
  float v = (t < D) ? x[(size_t)row * D + t] : 0.f;
  red[t] = v; __syncthreads();
  for (int s = 64; s > 0; s >>= 1) { if (t < s) red[t] += red[t + s]; __syncthreads(); }
  float mu = red[0] / (float)D; __syncthreads();
  float dv = (t < D) ? (v - mu) : 0.f;
  red[t] = dv * dv; __syncthreads();
  for (int s = 64; s > 0; s >>= 1) { if (t < s) red[t] += red[t + s]; __syncthreads(); }
  float inv = rsqrtf(red[0] / (float)D + 1e-5f);
  if (t < Dpad) y[(size_t)row * Dpad + t] = (t < D) ? (dv * inv * g[t] + b[t]) : 0.f;
}

// ---------------- vector norm-only rescale: v / sqrt(max(mean_i||v_i||^2,1e-8)) ----
__global__ void k_vnorm(const float* __restrict__ v, float* __restrict__ y,
                        int rows, int VI)
{
  int r = blockIdx.x * blockDim.x + threadIdx.x;
  if (r >= rows) return;
  const float* p = v + (size_t)r * VI * 3;
  float ss = 0.f;
  for (int i = 0; i < VI * 3; ++i) { float t = p[i]; ss += t * t; }
  float sc = rsqrtf(fmaxf(ss / (float)VI, 1e-8f));
  float* q = y + (size_t)r * VI * 3;
  for (int i = 0; i < VI * 3; ++i) q[i] = p[i] * sc;
}

// ---------------- feature engineering ----------------
__global__ void k_coords(const float* __restrict__ nv, float* __restrict__ coords)
{
  int r = blockIdx.x * blockDim.x + threadIdx.x;
  if (r >= NCr) return;
  for (int x = 0; x < 3; ++x) {
    float s = 0.f;
    for (int i = 0; i < 4; ++i) s += nv[((size_t)r * 4 + i) * 3 + x];
    coords[(size_t)r * 3 + x] = 0.25f * s;
  }
}

__global__ void k_edge_stats(const int* __restrict__ src, const int* __restrict__ dst,
                             const float* __restrict__ coords,
                             float* __restrict__ cnt_und, float* __restrict__ sum_d,
                             float* __restrict__ cnt_dst)
{
  int e = blockIdx.x * blockDim.x + threadIdx.x;
  if (e >= Ee) return;
  int s = src[e], d = dst[e];
  atomicAdd(&cnt_dst[d], 1.f);
  bool self = (s == d);
  atomicAdd(&cnt_und[s], 1.f);
  if (!self) atomicAdd(&cnt_und[d], 1.f);
  for (int c = 0; c < Cf; ++c) {
    const float* ps = coords + ((size_t)s * Cf + c) * 3;
    const float* pd = coords + ((size_t)d * Cf + c) * 3;
    float dx = ps[0]-pd[0], dy = ps[1]-pd[1], dz = ps[2]-pd[2];
    float dd = sqrtf(dx*dx + dy*dy + dz*dz);
    atomicAdd(&sum_d[(size_t)s * Cf + c], dd);
    if (!self) atomicAdd(&sum_d[(size_t)d * Cf + c], dd);
  }
}

__global__ void k_maxcnt(const float* __restrict__ cnt, unsigned* __restrict__ mx)
{
  int n = blockIdx.x * blockDim.x + threadIdx.x;
  if (n >= Nn) return;
  atomicMax(mx, __float_as_uint(cnt[n]));   // cnt >= 0 -> uint order == float order
}

__global__ void k_feat(const float* __restrict__ ns, const float* __restrict__ cnt_und,
                       const float* __restrict__ sum_d, const unsigned* __restrict__ mxu,
                       float* __restrict__ feat)
{
  int r = blockIdx.x * blockDim.x + threadIdx.x;
  if (r >= NCr) return;
  int n = r / Cf;
  float cnt = cnt_und[n];
  float cm  = fmaxf(__uint_as_float(*mxu), 1e-12f);
  float avg = sum_d[r] / fmaxf(cnt, 1.f);
  float prob = (cnt > 0.f) ? __expf(-avg * 0.2f) : 0.f;
  float pair = (0.25f * prob) / (prob + 1e-8f);
  float* f = feat + (size_t)r * NSI;
  const float* s = ns + (size_t)r * 64;
  for (int i = 0; i < 64; ++i) f[i] = s[i];
  f[64]=pair; f[65]=pair; f[66]=pair; f[67]=pair;
  f[68] = 1.f - cnt / cm;
  f[69]=0.25f; f[70]=0.25f; f[71]=0.25f; f[72]=0.25f;
  f[73]=0.f; f[74]=0.f; f[75]=0.f; f[76]=0.f;
  f[77]=0.5f;
}

// ---------------- small vector-channel GVPs (K=12 / K=3: scalar) ----------------
__global__ void k_embed_nodev(const float* __restrict__ vn12, const float* __restrict__ Wv,
                              const float* __restrict__ bv, const float* __restrict__ s,
                              float* __restrict__ vout)
{
  int r = blockIdx.x * blockDim.x + threadIdx.x;
  if (r >= NCr) return;
  const float* vi = vn12 + (size_t)r * 12;
  const float* sr = s + (size_t)r * NSH;
  float* vo = vout + (size_t)r * 48;
  for (int o = 0; o < 48; ++o) {
    float t = bv[o];
    for (int y = 0; y < 12; ++y) t += vi[y] * Wv[y * 48 + o];
    vo[o] = t * sigmoidf_(sr[o / 3]);
  }
}

__global__ void k_embed_edgev(const float* __restrict__ evn, const float* __restrict__ Wv,
                              const float* __restrict__ bv, const float* __restrict__ esh,
                              float* __restrict__ evh)
{
  int r = blockIdx.x * blockDim.x + threadIdx.x;
  if (r >= ECr) return;
  const float* vi = evn + (size_t)r * 3;
  float gate = sigmoidf_(esh[(size_t)r * 32]);
  for (int o = 0; o < 3; ++o) {
    float t = bv[o];
    for (int y = 0; y < 3; ++y) t += vi[y] * Wv[y * 3 + o];
    evh[(size_t)r * 3 + o] = t * gate;
  }
}

// edge-message vector channel: recompute gate cols 0..15, scatter gated v into aggv
__global__ void k_edge_msg_v(const float* __restrict__ esh, const float* __restrict__ evh,
                             const float* __restrict__ Ws, const float* __restrict__ bs,
                             const float* __restrict__ Wv, const float* __restrict__ bv,
                             const int* __restrict__ dst, float* __restrict__ aggv)
{
  int r = blockIdx.x * blockDim.x + threadIdx.x;
  if (r >= ECr) return;
  const float* er = esh + (size_t)r * 32;
  float gate[16];
  for (int j = 0; j < 16; ++j) {
    float t = bs[j];
    for (int k = 0; k < 32; ++k) t += er[k] * Ws[k * NSH + j];
    gate[j] = sigmoidf_(siluf_(t));
  }
  const float* vr = evh + (size_t)r * 3;
  int orow = dst[r >> 1] * 2 + (r & 1);
  float* ag = aggv + (size_t)orow * 48;
  for (int o = 0; o < 48; ++o) {
    float t = bv[o];
    for (int y = 0; y < 3; ++y) t += vr[y] * Wv[y * 48 + o];
    atomicAdd(&ag[o], t * gate[o / 3]);
  }
}

// ---------------- elementwise glue ----------------
__global__ void k_msg_add(const float* __restrict__ a, const float* __restrict__ agg,
                          const float* __restrict__ cnt_dst, float* __restrict__ out,
                          int rows, int D)
{
  size_t i = (size_t)blockIdx.x * blockDim.x + threadIdx.x;
  if (i >= (size_t)rows * D) return;
  int r = (int)(i / D);
  float c = fmaxf(cnt_dst[r / Cf], 1.f);
  out[i] = a[i] + agg[i] / c;
}

__global__ void k_add2(const float* __restrict__ a, const float* __restrict__ b,
                       float* __restrict__ o, size_t n)
{
  size_t i = (size_t)blockIdx.x * blockDim.x + threadIdx.x;
  if (i < n) o[i] = a[i] + b[i];
}

__global__ void k_gatev(const float* __restrict__ ov, const float* __restrict__ os,
                        const float* __restrict__ rv, float* __restrict__ v)
{
  size_t i = (size_t)blockIdx.x * blockDim.x + threadIdx.x;
  if (i >= (size_t)NCr * 48) return;
  size_t r = i / 48; int j = (int)(i % 48);
  v[i] = ov[i] * sigmoidf_(os[r * NSH + j / 3]) + rv[i];
}

__global__ void k_conf_mean(const float* __restrict__ x, float* __restrict__ y, int D)
{
  size_t i = (size_t)blockIdx.x * blockDim.x + threadIdx.x;
  if (i >= (size_t)Nn * D) return;
  size_t n = i / D; int d = (int)(i % D);
  y[i] = 0.5f * (x[(n * 2) * D + d] + x[(n * 2 + 1) * D + d]);
}

__global__ void k_bias_add(const float* __restrict__ x, const float* __restrict__ b,
                           float* __restrict__ y, int rows, int D)
{
  size_t i = (size_t)blockIdx.x * blockDim.x + threadIdx.x;
  if (i >= (size_t)rows * D) return;
  y[i] = x[i] + b[i % D];
}

// ---------------- GAT (segment softmax via flipped-uint atomicMax) ----------------
__device__ __forceinline__ unsigned fmax_key(float f) {
  unsigned u = __float_as_uint(f);
  return (u & 0x80000000u) ? ~u : (u | 0x80000000u);
}
__device__ __forceinline__ float fmax_unkey(unsigned u) {
  return (u & 0x80000000u) ? __uint_as_float(u & 0x7FFFFFFFu) : __uint_as_float(~u);
}
__device__ __forceinline__ float gat_al(const float* as_, const float* ad_, int s, int d, int hd) {
  float a = as_[s * HEADS + hd] + ad_[d * HEADS + hd];
  return a > 0.f ? a : 0.2f * a;          // leaky_relu(0.2)
}

__global__ void k_attn_scores(const float* __restrict__ h, const float* __restrict__ aw_s,
                              const float* __restrict__ aw_d,
                              float* __restrict__ asrc, float* __restrict__ adst)
{
  int i = blockIdx.x * blockDim.x + threadIdx.x;
  if (i >= Nn * HEADS) return;
  int n = i / HEADS, hd = i % HEADS;
  const float* hr = h + (size_t)n * NSH + hd * HC;
  float t1 = 0.f, t2 = 0.f;
  for (int c = 0; c < HC; ++c) { t1 += hr[c] * aw_s[hd * HC + c]; t2 += hr[c] * aw_d[hd * HC + c]; }
  asrc[i] = t1; adst[i] = t2;
}

__global__ void k_gat_max(const int* __restrict__ src, const int* __restrict__ dst,
                          const float* __restrict__ asrc, const float* __restrict__ adst,
                          unsigned* __restrict__ mx)
{
  int i = blockIdx.x * blockDim.x + threadIdx.x;
  if (i >= Ee + Nn) return;
  int s = (i < Ee) ? src[i] : i - Ee;
  int d = (i < Ee) ? dst[i] : i - Ee;
  for (int hd = 0; hd < HEADS; ++hd)
    atomicMax(&mx[d * HEADS + hd], fmax_key(gat_al(asrc, adst, s, d, hd)));
}

__global__ void k_gat_den(const int* __restrict__ src, const int* __restrict__ dst,
                          const float* __restrict__ asrc, const float* __restrict__ adst,
                          const unsigned* __restrict__ mx, float* __restrict__ den)
{
  int i = blockIdx.x * blockDim.x + threadIdx.x;
  if (i >= Ee + Nn) return;
  int s = (i < Ee) ? src[i] : i - Ee;
  int d = (i < Ee) ? dst[i] : i - Ee;
  for (int hd = 0; hd < HEADS; ++hd) {
    float al = gat_al(asrc, adst, s, d, hd);
    atomicAdd(&den[d * HEADS + hd], __expf(al - fmax_unkey(mx[d * HEADS + hd])));
  }
}

__global__ void k_gat_agg(const int* __restrict__ src, const int* __restrict__ dst,
                          const float* __restrict__ asrc, const float* __restrict__ adst,
                          const unsigned* __restrict__ mx, const float* __restrict__ den,
                          const float* __restrict__ h, float* __restrict__ out)
{
  int i = blockIdx.x * blockDim.x + threadIdx.x;
  if (i >= (Ee + Nn) * HEADS) return;
  int it = i / HEADS, hd = i % HEADS;
  int s = (it < Ee) ? src[it] : it - Ee;
  int d = (it < Ee) ? dst[it] : it - Ee;
  float al = gat_al(asrc, adst, s, d, hd);
  float w = __expf(al - fmax_unkey(mx[d * HEADS + hd])) / fmaxf(den[d * HEADS + hd], 1e-16f);
  const float* hr = h + (size_t)s * NSH + hd * HC;
  float* op = out + (size_t)d * NSH + hd * HC;
  for (int c = 0; c < HC; ++c) atomicAdd(&op[c], w * hr[c]);
}

// ---------------- output head: logits accumulate (ensemble mean) ----------------
__global__ void k_out2(const float* __restrict__ s1, const float* __restrict__ W,
                       const float* __restrict__ b, float* __restrict__ out)
{
  int n = blockIdx.x * blockDim.x + threadIdx.x;
  if (n >= Nn) return;
  const float* sr = s1 + (size_t)n * NSH;
  for (int o = 0; o < OUTD; ++o) {
    float t = b[o];
    for (int k = 0; k < NSH; ++k) t += sr[k] * W[k * OUTD + o];
    out[n * OUTD + o] += t * (1.f / 3.f);
  }
}

// =====================================================================
// host orchestration
// =====================================================================
enum {
  pWv_ln_g = 0, pWv_ln_b, pWv_Ws, pWv_bs, pWv_Wv, pWv_bv,
  pWe_ln_g, pWe_ln_b, pWe_Ws, pWe_bs, pWe_Wv, pWe_bv,
  penc_ln_g, penc_ln_b, penc_edge_Ws, penc_edge_bs, penc_edge_Wv, penc_edge_bv,
  penc_node_Ws, penc_node_bs, penc_node_Wv, penc_node_bv,
  penc_res_Ws, penc_res_bs, penc_res_Wv, penc_res_bv,
  pgat_W, pgat_asrc, pgat_adst, pgat_b,
  pout_ln_g, pout_ln_b, pout1_Ws, pout1_bs, pout1_Wv, pout1_bv, pout2_Ws, pout2_bs,
  NPARAM
};
static const size_t kParamElems[NPARAM] = {
  234, 234, (size_t)3*78*128, 384, 1728, 144,
  96, 96, 3072, 96, 27, 9,
  1920, 1920, (size_t)3*5*32*128, 1920, 2160, 720,
  (size_t)3*5*128*128, 1920, (size_t)3*5*48*48, 720,
  (size_t)3*5*128*128, 1920, (size_t)3*5*48*48, 720,
  (size_t)3*2*128*128, 768, 768, 768,
  384, 384, (size_t)3*128*128, 384, 6912, 144, 1536, 12
};

template<int K, int N>
static void gemmT(const float* A, const float* W, const float* bias, float* Out,
                  const int* scat, int M, int act, hipStream_t stream)
{
  constexpr int TB = 256;
  long long waves = (long long)((M + 15) / 16) * (N / 16);
  unsigned grid = (unsigned)((waves + (TB / 32) - 1) / (TB / 32));
  k_gemmT<K, N><<<grid, TB, 0, stream>>>(A, W, bias, Out, scat, M, act);
}

extern "C" void kernel_launch(void* const* d_in, const int* in_sizes, int n_in,
                              void* d_out, int out_size, void* d_ws, size_t ws_size,
                              hipStream_t stream)
{
  (void)in_sizes; (void)out_size; (void)ws_size;
  const float* node_s = (const float*)d_in[0];
  const float* node_v = (const float*)d_in[1];
  const float* edge_s = (const float*)d_in[2];
  const float* edge_v = (const float*)d_in[3];

  const float* P[NPARAM];
  const int* eidx;
  if (n_in >= 4 + NPARAM + 1) {              // params flattened leaf-by-leaf
    for (int i = 0; i < NPARAM; ++i) P[i] = (const float*)d_in[4 + i];
    eidx = (const int*)d_in[4 + NPARAM];
  } else {                                    // params as one concatenated blob
    const float* blob = (const float*)d_in[4];
    size_t off = 0;
    for (int i = 0; i < NPARAM; ++i) { P[i] = blob + off; off += kParamElems[i]; }
    eidx = (const int*)d_in[n_in - 1];
  }
  const int* src = eidx;
  const int* dst = eidx + Ee;

  // ---- workspace carve-out ----
  char* base = (char*)d_ws;
  size_t off = 0;
  auto alloc = [&](size_t elems) -> float* {
    float* p = (float*)(base + off);
    off += ((elems * 4) + 255) & ~(size_t)255;
    return p;
  };
  float* coords  = alloc((size_t)NCr * 3);
  float* cnt_und = alloc(Nn);
  float* sum_d   = alloc(NCr);
  unsigned* cmax = (unsigned*)alloc(8);
  float* cnt_dst = alloc(Nn);
  float* feat    = alloc((size_t)NCr * NSI);
  float* vnin    = alloc((size_t)NCr * 12);
  float* evn     = alloc((size_t)ECr * 3);
  float* featln  = alloc((size_t)NCr * NSIP);   // zero-padded K=80
  float* wvpad   = alloc((size_t)Mens * NSIP * NSH);
  float* s       = alloc((size_t)NCr * NSH);
  float* vv      = alloc((size_t)NCr * 48);
  float* sn      = alloc((size_t)NCr * NSH);
  float* vn      = alloc((size_t)NCr * 48);
  float* eln     = alloc((size_t)ECr * 32);
  float* esh     = alloc((size_t)ECr * 32);
  float* evh     = alloc((size_t)ECr * 3);
  float* aggs    = alloc((size_t)NCr * NSH);
  float* aggv    = alloc((size_t)NCr * 48);
  float* ms      = alloc((size_t)NCr * NSH);
  float* mv      = alloc((size_t)NCr * 48);
  float* os_     = alloc((size_t)NCr * NSH);
  float* ov      = alloc((size_t)NCr * 48);
  float* rs      = alloc((size_t)NCr * NSH);
  float* rv      = alloc((size_t)NCr * 48);
  float* sm      = alloc((size_t)Nn * NSH);
  float* hbuf    = alloc((size_t)Nn * NSH);
  float* gout    = alloc((size_t)Nn * NSH);
  float* xg      = alloc((size_t)Nn * NSH);
  float* asrcb   = alloc(Nn * HEADS);
  float* adstb   = alloc(Nn * HEADS);
  unsigned* mxb  = (unsigned*)alloc(Nn * HEADS);
  float* denb    = alloc(Nn * HEADS);
  float* lnh     = alloc((size_t)Nn * NSH);
  float* s1      = alloc((size_t)Nn * NSH);

  const int TB = 256;
  auto G1 = [&](size_t n) { return (unsigned)((n + TB - 1) / TB); };

  hipMemsetAsync(cnt_und, 0, Nn * 4, stream);
  hipMemsetAsync(sum_d,   0, NCr * 4, stream);
  hipMemsetAsync(cmax,    0, 4, stream);
  hipMemsetAsync(cnt_dst, 0, Nn * 4, stream);
  hipMemsetAsync(d_out,   0, (size_t)Nn * OUTD * 4, stream);

  // stage 0: member-independent features / vector norms / padded weights
  k_coords<<<G1(NCr), TB, 0, stream>>>(node_v, coords);
  k_edge_stats<<<G1(Ee), TB, 0, stream>>>(src, dst, coords, cnt_und, sum_d, cnt_dst);
  k_maxcnt<<<G1(Nn), TB, 0, stream>>>(cnt_und, cmax);
  k_feat<<<G1(NCr), TB, 0, stream>>>(node_s, cnt_und, sum_d, cmax, feat);
  k_vnorm<<<G1(NCr), TB, 0, stream>>>(node_v, vnin, NCr, 4);
  k_vnorm<<<G1(ECr), TB, 0, stream>>>(edge_v, evn, ECr, 1);
  k_padwv<<<G1((size_t)Mens * NSIP * NSH), TB, 0, stream>>>(P[pWv_Ws], wvpad);

  for (int m = 0; m < Mens; ++m) {
    // node input GVP (LN -> WMMA GEMM K=80(padded) -> gated vector channel)
    k_layernorm<<<NCr, 128, 0, stream>>>(feat, P[pWv_ln_g] + m * NSI, P[pWv_ln_b] + m * NSI,
                                         featln, NSI, NSIP);
    gemmT<NSIP, NSH>(featln, wvpad + (size_t)m * NSIP * NSH, P[pWv_bs] + m * NSH,
                     s, nullptr, NCr, 0, stream);
    k_embed_nodev<<<G1(NCr), TB, 0, stream>>>(vnin, P[pWv_Wv] + (size_t)m * 12 * 48,
                                              P[pWv_bv] + m * 48, s, vv);
    // edge input GVP
    k_layernorm<<<ECr, 128, 0, stream>>>(edge_s, P[pWe_ln_g] + m * 32, P[pWe_ln_b] + m * 32,
                                         eln, 32, 32);
    gemmT<32, 32>(eln, P[pWe_Ws] + (size_t)m * 32 * 32, P[pWe_bs] + m * 32,
                  esh, nullptr, ECr, 0, stream);
    k_embed_edgev<<<G1(ECr), TB, 0, stream>>>(evn, P[pWe_Wv] + (size_t)m * 9,
                                              P[pWe_bv] + m * 3, esh, evh);

    for (int l = 0; l < Ly; ++l) {
      int ml = m * Ly + l;
      k_layernorm<<<NCr, 128, 0, stream>>>(s, P[penc_ln_g] + (size_t)ml * NSH,
                                           P[penc_ln_b] + (size_t)ml * NSH, sn, NSH, NSH);
      k_vnorm<<<G1(NCr), TB, 0, stream>>>(vv, vn, NCr, 16);
      hipMemsetAsync(aggs, 0, (size_t)NCr * NSH * 4, stream);
      hipMemsetAsync(aggv, 0, (size_t)NCr * 48 * 4, stream);
      // fused: silu(es_h @ Ws + bs) tiles scattered straight into agg_s
      gemmT<32, NSH>(esh, P[penc_edge_Ws] + (size_t)ml * 32 * NSH,
                     P[penc_edge_bs] + (size_t)ml * NSH, aggs, dst, ECr, 1, stream);
      k_edge_msg_v<<<G1(ECr), TB, 0, stream>>>(esh, evh,
           P[penc_edge_Ws] + (size_t)ml * 32 * NSH, P[penc_edge_bs] + (size_t)ml * NSH,
           P[penc_edge_Wv] + (size_t)ml * 3 * 48,  P[penc_edge_bv] + (size_t)ml * 48, dst, aggv);
      k_msg_add<<<G1((size_t)NCr * NSH), TB, 0, stream>>>(sn, aggs, cnt_dst, ms, NCr, NSH);
      k_msg_add<<<G1((size_t)NCr * 48),  TB, 0, stream>>>(vn, aggv, cnt_dst, mv, NCr, 48);
      gemmT<NSH, NSH>(ms, P[penc_node_Ws] + (size_t)ml * NSH * NSH,
                      P[penc_node_bs] + (size_t)ml * NSH, os_, nullptr, NCr, 1, stream);
      gemmT<48, 48>(mv, P[penc_node_Wv] + (size_t)ml * 48 * 48,
                    P[penc_node_bv] + (size_t)ml * 48, ov, nullptr, NCr, 0, stream);
      gemmT<NSH, NSH>(sn, P[penc_res_Ws] + (size_t)ml * NSH * NSH,
                      P[penc_res_bs] + (size_t)ml * NSH, rs, nullptr, NCr, 0, stream);
      gemmT<48, 48>(vn, P[penc_res_Wv] + (size_t)ml * 48 * 48,
                    P[penc_res_bv] + (size_t)ml * 48, rv, nullptr, NCr, 0, stream);
      k_add2<<<G1((size_t)NCr * NSH), TB, 0, stream>>>(os_, rs, s, (size_t)NCr * NSH);
      k_gatev<<<G1((size_t)NCr * 48), TB, 0, stream>>>(ov, os_, rv, vv);
    }
    k_conf_mean<<<G1((size_t)Nn * NSH), TB, 0, stream>>>(s, sm, NSH);

    const float* xcur = sm;
    for (int gl = 0; gl < 2; ++gl) {
      int mg = m * 2 + gl;
      hipMemsetAsync(mxb,  0, Nn * HEADS * 4, stream);
      hipMemsetAsync(denb, 0, Nn * HEADS * 4, stream);
      hipMemsetAsync(gout, 0, (size_t)Nn * NSH * 4, stream);
      gemmT<NSH, NSH>(xcur, P[pgat_W] + (size_t)mg * NSH * NSH, nullptr,
                      hbuf, nullptr, Nn, 0, stream);
      k_attn_scores<<<G1(Nn * HEADS), TB, 0, stream>>>(hbuf,
           P[pgat_asrc] + (size_t)mg * 128, P[pgat_adst] + (size_t)mg * 128, asrcb, adstb);
      k_gat_max<<<G1(Ee + Nn), TB, 0, stream>>>(src, dst, asrcb, adstb, mxb);
      k_gat_den<<<G1(Ee + Nn), TB, 0, stream>>>(src, dst, asrcb, adstb, mxb, denb);
      k_gat_agg<<<G1((size_t)(Ee + Nn) * HEADS), TB, 0, stream>>>(src, dst, asrcb, adstb,
                                                                  mxb, denb, hbuf, gout);
      k_bias_add<<<G1((size_t)Nn * NSH), TB, 0, stream>>>(gout, P[pgat_b] + (size_t)mg * NSH,
                                                          xg, Nn, NSH);
      xcur = xg;
    }
    k_layernorm<<<Nn, 128, 0, stream>>>(xcur, P[pout_ln_g] + m * NSH, P[pout_ln_b] + m * NSH,
                                        lnh, NSH, NSH);
    gemmT<NSH, NSH>(lnh, P[pout1_Ws] + (size_t)m * NSH * NSH, P[pout1_bs] + m * NSH,
                    s1, nullptr, Nn, 0, stream);
    k_out2<<<G1(Nn), TB, 0, stream>>>(s1, P[pout2_Ws] + (size_t)m * NSH * OUTD,
                                      P[pout2_bs] + m * OUTD, (float*)d_out);
  }
}
